// ChebyKANLinear_57131654972007
// MI455X (gfx1250) — matrix operand order repaired and checked by reading, not compiled, not run on hardware
//
#include <hip/hip_runtime.h>
#include <hip/hip_bf16.h>

// ---------------------------------------------------------------------------
// ChebyKAN linear layer for MI455X (gfx1250, wave32, WMMA).
//
//   y[b,o] = (1/I) * sum_{i,d} T_d(clip(tanh(x[b,i]))) * C[i,o,d]
//
// Reformulated as GEMM:  Y[2048,256] = Basis[2048,1024] x W[1024,256]
// with k = i*4 + d.  WMMA path: v_wmma_f32_16x16x32_bf16 with split-bf16
// (hi+lo) inputs for ~fp32 accuracy:  C += Ah*Bh + Ah*Bl + Al*Bh.
// ---------------------------------------------------------------------------

typedef __bf16 bf16_t;
typedef bf16_t bf16x16 __attribute__((ext_vector_type(16)));
typedef float  f32x8   __attribute__((ext_vector_type(8)));

#define BATCH 2048
#define IDIM  256
#define ODIM  256
#define DEG1  4
#define KDIM  (IDIM * DEG1)   // 1024
#define MT    (BATCH / 16)    // 128 M-tiles
#define NT    (ODIM / 16)     // 16  N-tiles
#define KT    (KDIM / 32)     // 32  K-tiles

// Fragment counts
#define A_FRAGS ((size_t)MT * KT)   // 4096 fragments * 32 lanes * 16 bf16
#define B_FRAGS ((size_t)KT * NT)   // 512

__device__ __forceinline__ void split_bf16(float v, bf16_t& hi, bf16_t& lo) {
    hi = (bf16_t)v;
    lo = (bf16_t)(v - (float)hi);
}

// ---------------------------------------------------------------------------
// Pack A = Chebyshev basis, directly in WMMA 16-bit A-fragment layout.
// Contiguous k order used here, not the ISA table order (16-bit A 16x32): lane l (g = l/16, m = l%16) holds, as 16
// bf16 elements j=0..15:  k_local = j + 16*g (matches B packing).
// With k = i*4 + d, each lane needs i in {base, base+1, base+2, base+3},
// base = kt*8 + 4*g, all four degrees each.
// One thread = one lane of one (mtile, ktile) fragment.
// ---------------------------------------------------------------------------
__global__ __launch_bounds__(256) void pack_basis(const float* __restrict__ x,
                                                  bf16_t* __restrict__ a_hi,
                                                  bf16_t* __restrict__ a_lo) {
    int tid  = blockIdx.x * 256 + threadIdx.x;     // MT*KT*32 threads
    int lane = tid & 31;
    int kt   = (tid >> 5) & (KT - 1);
    int mt   = tid >> 10;                          // / (32*32)
    int g    = lane >> 4;
    int row  = mt * 16 + (lane & 15);              // batch index b
    int ibas = kt * 8 + g * 4;
    const int ioff[4] = {0, 1, 2, 3};

    bf16x16 fh, fl;
#pragma unroll
    for (int q = 0; q < 4; ++q) {
        int i   = ibas + ioff[q];
        float t = tanhf(x[row * IDIM + i]);
        t = fminf(fmaxf(t, -1.0f + 1e-7f), 1.0f - 1e-7f);
        float T0 = 1.0f;
        float T1 = t;
        float T2 = 2.0f * t * t - 1.0f;
        float T3 = 2.0f * t * T2 - T1;
        bf16_t h, l;
        split_bf16(T0, h, l); fh[q * 4 + 0] = h; fl[q * 4 + 0] = l;
        split_bf16(T1, h, l); fh[q * 4 + 1] = h; fl[q * 4 + 1] = l;
        split_bf16(T2, h, l); fh[q * 4 + 2] = h; fl[q * 4 + 2] = l;
        split_bf16(T3, h, l); fh[q * 4 + 3] = h; fl[q * 4 + 3] = l;
    }
    size_t idx = ((size_t)(mt * KT + kt) * 32 + lane);
    ((bf16x16*)a_hi)[idx] = fh;
    ((bf16x16*)a_lo)[idx] = fl;
}

// ---------------------------------------------------------------------------
// Pack B = coeffs (scaled by 1/IDIM, a power of two -> lossless pre-bf16)
// in WMMA 16-bit B-fragment layout (32x16 K x N per tile):
// lane l (h = l/16, col o = l%16) holds K = kt*32 + 16*h + j, j = 0..15.
// k -> (i = k/4, d = k%4): four consecutive i, float4 gather per (i,o).
// ---------------------------------------------------------------------------
__global__ __launch_bounds__(256) void pack_coeffs(const float* __restrict__ w,
                                                   bf16_t* __restrict__ b_hi,
                                                   bf16_t* __restrict__ b_lo) {
    int tid  = blockIdx.x * 256 + threadIdx.x;     // KT*NT*32 threads
    int lane = tid & 31;
    int nt   = (tid >> 5) & (NT - 1);
    int kt   = tid >> 9;                           // / (NT*32)
    int h    = lane >> 4;
    int o    = nt * 16 + (lane & 15);
    int ibas = kt * 8 + h * 4;
    const float s = 1.0f / (float)IDIM;

    bf16x16 fh, fl;
#pragma unroll
    for (int q = 0; q < 4; ++q) {
        int i = ibas + q;
        const float4 c = ((const float4*)w)[i * ODIM + o];  // C[i,o,0..3]
        bf16_t hh, ll;
        split_bf16(c.x * s, hh, ll); fh[q * 4 + 0] = hh; fl[q * 4 + 0] = ll;
        split_bf16(c.y * s, hh, ll); fh[q * 4 + 1] = hh; fl[q * 4 + 1] = ll;
        split_bf16(c.z * s, hh, ll); fh[q * 4 + 2] = hh; fl[q * 4 + 2] = ll;
        split_bf16(c.w * s, hh, ll); fh[q * 4 + 3] = hh; fl[q * 4 + 3] = ll;
    }
    size_t idx = ((size_t)(kt * NT + nt) * 32 + lane);
    ((bf16x16*)b_hi)[idx] = fh;
    ((bf16x16*)b_lo)[idx] = fl;
}

// ---------------------------------------------------------------------------
// GEMM: each wave owns a 64(M) x 32(N) macro-tile = 4x2 WMMA sub-tiles.
// 256 waves total (32 M-supertiles x 8 N-supertiles), 8 waves per block.
// Per K-tile: 12 fragment loads (b128 pairs) + 24 v_wmma_f32_16x16x32_bf16.
// Split accumulation: C += Ah*Bh + Ah*Bl + Al*Bh  (drop lo*lo, ~2^-16 rel).
// ---------------------------------------------------------------------------
__global__ __launch_bounds__(256) void cheby_gemm(const bf16_t* __restrict__ a_hi_p,
                                                  const bf16_t* __restrict__ a_lo_p,
                                                  const bf16_t* __restrict__ b_hi_p,
                                                  const bf16_t* __restrict__ b_lo_p,
                                                  float* __restrict__ y) {
    const bf16x16* AH = (const bf16x16*)a_hi_p;
    const bf16x16* AL = (const bf16x16*)a_lo_p;
    const bf16x16* BH = (const bf16x16*)b_hi_p;
    const bf16x16* BL = (const bf16x16*)b_lo_p;

    int lane = threadIdx.x & 31;
    int wave = (blockIdx.x * blockDim.x + threadIdx.x) >> 5;  // 0..255
    int wn   = wave & 7;        // N-supertile (2 N-tiles)
    int wm   = wave >> 3;       // M-supertile (4 M-tiles)
    int mt0  = wm * 4;
    int nt0  = wn * 2;

    f32x8 acc[4][2];
#pragma unroll
    for (int im = 0; im < 4; ++im)
#pragma unroll
        for (int in = 0; in < 2; ++in)
            acc[im][in] = {};

#pragma unroll 1
    for (int kt = 0; kt < KT; ++kt) {
        bf16x16 ah[4], al[4], bh[2], bl[2];
#pragma unroll
        for (int im = 0; im < 4; ++im) {
            size_t ai = ((size_t)((mt0 + im) * KT + kt) * 32 + lane);
            ah[im] = AH[ai];
            al[im] = AL[ai];
        }
#pragma unroll
        for (int in = 0; in < 2; ++in) {
            size_t bi = ((size_t)(kt * NT + nt0 + in) * 32 + lane);
            bh[in] = BH[bi];
            bl[in] = BL[bi];
        }
#pragma unroll
        for (int im = 0; im < 4; ++im) {
#pragma unroll
            for (int in = 0; in < 2; ++in) {
                // args: (neg_a, A, neg_b, B, c_mod, C, reuse_a, reuse_b)
                acc[im][in] = __builtin_amdgcn_wmma_f32_16x16x32_bf16(
                    false, ah[im], false, bh[in], (short)0, acc[im][in], false, false);
                acc[im][in] = __builtin_amdgcn_wmma_f32_16x16x32_bf16(
                    false, ah[im], false, bl[in], (short)0, acc[im][in], false, false);
                acc[im][in] = __builtin_amdgcn_wmma_f32_16x16x32_bf16(
                    false, al[im], false, bh[in], (short)0, acc[im][in], false, false);
            }
        }
    }

    // C/D layout (ISA 7.12.2): VGPR r, lane L -> M = r + 8*(L/16), N = L%16.
    int rbase = 8 * (lane >> 4);
    int col   = lane & 15;
#pragma unroll
    for (int im = 0; im < 4; ++im) {
#pragma unroll
        for (int in = 0; in < 2; ++in) {
            int o = (nt0 + in) * 16 + col;
#pragma unroll
            for (int r = 0; r < 8; ++r) {
                int b = (mt0 + im) * 16 + rbase + r;
                y[b * ODIM + o] = acc[im][in][r];
            }
        }
    }
}

// ---------------------------------------------------------------------------
extern "C" void kernel_launch(void* const* d_in, const int* in_sizes, int n_in,
                              void* d_out, int out_size, void* d_ws, size_t ws_size,
                              hipStream_t stream) {
    const float* x = (const float*)d_in[0];        // [2048, 256]
    const float* w = (const float*)d_in[1];        // [256, 256, 4]
    float* y       = (float*)d_out;                // [2048, 256]

    // Workspace layout (9 MB total): A_hi | A_lo | B_hi | B_lo
    bf16_t* a_hi = (bf16_t*)d_ws;
    bf16_t* a_lo = a_hi + (size_t)BATCH * KDIM;    // +4 MB
    bf16_t* b_hi = a_lo + (size_t)BATCH * KDIM;    // +4 MB
    bf16_t* b_lo = b_hi + (size_t)KDIM * ODIM;     // +512 KB

    pack_basis <<<(MT * KT * 32) / 256, 256, 0, stream>>>(x, a_hi, a_lo);
    pack_coeffs<<<(KT * NT * 32) / 256, 256, 0, stream>>>(w, b_hi, b_lo);
    cheby_gemm <<<(256 * 32) / 256, 256, 0, stream>>>(a_hi, a_lo, b_hi, b_lo, y);
}
